// hyperRNN_7172595385064
// MI455X (gfx1250) — matrix operand order repaired and checked by reading, not compiled, not run on hardware
//
#include <hip/hip_runtime.h>
#include <hip/hip_bf16.h>

// ---------------------------------------------------------------------------
// Hyperbolic RNN (Mobius scan) for gfx1250 / MI455X — two-phase decomposition.
//   inputs: x[B,S,H,d] f32, w[Hd,Hd] f32, u[Hd,Hd] f32, b[Hd] f32
//   out:    h[B,S,H,d] f32
//
// Phase 1 (chip-saturating, 16384 WGs): Ux' = mob_mat_mul(u, x) for ALL (b,s)
//   positions — independent of the recurrence. [B*S,128] @ [128,128] via
//   v_wmma_f32_16x16x32_f16 + per-head tanh/atanh tail. Result written into
//   d_out (same shape), consumed in-place by phase 2.
// Phase 2 (serial scan, 32 WGs x 512 thr): h <- mob_add(mob_add(W(*)h, Ux'), b)
//   with only the W GEMM on the critical path, K-split across wave pairs,
//   weights register-resident, split barriers, Ux'(t+1) pipelined in regs.
// ---------------------------------------------------------------------------

typedef __attribute__((ext_vector_type(16))) _Float16 v16h;
typedef __attribute__((ext_vector_type(8)))  float    v8f;

constexpr int   kB = 512, kS = 512, kH = 64, kHD = 128;
constexpr int   kBT = 16;           // rows (batch or flat) per workgroup tile
constexpr int   kP1Threads = 256;   // 8 waves
constexpr int   kP2Threads = 512;   // 16 waves
constexpr float kEps = 1e-15f;
constexpr float kProjEps = 1e-5f;

#if __has_builtin(__builtin_amdgcn_tanhf)
#define FAST_TANH(x) __builtin_amdgcn_tanhf(x)   // native v_tanh_f32 on gfx1250
#else
#define FAST_TANH(x) tanhf(x)
#endif

// Split workgroup barrier: LDS stores made visible, then signal; independent
// register-only / global-load work may execute between SIGNAL and WAIT.
#define BAR_SIGNAL() asm volatile("s_wait_dscnt 0x0\n\ts_barrier_signal -1" ::: "memory")
#define BAR_WAIT()   asm volatile("s_barrier_wait -1" ::: "memory")

__device__ __forceinline__ float frcp(float v) {
#if __has_builtin(__builtin_amdgcn_rcpf)
    return __builtin_amdgcn_rcpf(v);             // v_rcp_f32, ~1 ulp
#else
    return 1.0f / v;
#endif
}

__device__ __forceinline__ float fsqrt(float v) {
#if __has_builtin(__builtin_amdgcn_sqrtf)
    return __builtin_amdgcn_sqrtf(v);            // v_sqrt_f32, ~1 ulp
#else
    return sqrtf(v);
#endif
}

__device__ __forceinline__ float2 project2(float x, float y) {
    float n2 = x * x + y * y;
    float s  = (1.0f - kProjEps) * frcp(fmaxf(n2, 1.0f - kProjEps));
    return make_float2(x * s, y * s);
}

__device__ __forceinline__ float2 mob_add2(float2 u, float2 v) {
    v.x += kEps; v.y += kEps;
    float nuv = 2.0f * (u.x * v.x + u.y * v.y);
    float nu  = u.x * u.x + u.y * u.y;
    float nv  = v.x * v.x + v.y * v.y;
    float rden = frcp(1.0f + nuv + nv * nu);
    float cu  = (1.0f + nuv + nv) * rden;
    float cv  = (1.0f - nu) * rden;
    return project2(cu * u.x + cv * v.x, cu * u.y + cv * v.y);
}

// mob_mat_mul tail: project( tanh(||Mx+eps||/xn * atanh(xn)) / ||Mx+eps|| * Mx )
__device__ __forceinline__ float2 mobmm_tail(float mx, float my, float xn) {
    float ax  = mx + kEps, ay = my + kEps;
    float Mxn = fsqrt(ax * ax + ay * ay);
    float at  = 0.5f * __logf((1.0f + xn) * frcp(1.0f - xn));  // atanh(xn)
    float r   = FAST_TANH(Mxn * frcp(xn) * at) * frcp(Mxn);
    return project2(r * mx, r * my);
}

// 16-bit A-matrix 16x32 fragment position (ISA 7.12.2):
//   lanes 0-15 hold M=lane: K 0..7 (j=0..7), K 16..23 (j=8..15)
//   lanes 16-31 hold M=lane-16: K 8..15 (j=0..7), K 24..31 (j=8..15)
__device__ __forceinline__ void frag_pos(int M, int K, int& kb, int& ln, int& j) {
    kb = K >> 5;
    int kl = K & 31;
    if (kl < 8)       { ln = M;      j = kl;      }
    else if (kl < 16) { ln = M + 16; j = kl - 8;  }
    else if (kl < 24) { ln = M;      j = kl - 8;  }
    else              { ln = M + 16; j = kl - 16; }
}

// ===========================================================================
// Phase 1: Ux' for every (b, s) — fully parallel, saturates the device.
// ===========================================================================
__global__ __launch_bounds__(kP1Threads)
void ux_precompute_kernel(const float* __restrict__ x,
                          const float* __restrict__ u,
                          float* __restrict__ out) {
    __shared__ _Float16 xfrag[4][32][16];
    __shared__ float UxS[kBT][kHD];
    __shared__ float xnS[kBT][kH];

    const int tid  = threadIdx.x;
    const int wave = tid >> 5;               // 0..7 -> N-tile
    const int lane = tid & 31;
    const size_t m0 = (size_t)blockIdx.x * kBT;   // flat row = b*S + s

    // Register-resident U B-fragments (one N-tile per wave).
    const int N = wave * 16 + (lane & 15);
    v16h Ub[4];
#pragma unroll
    for (int kb = 0; kb < 4; ++kb) {
#pragma unroll
        for (int j = 0; j < 16; ++j) {
            const int K = kb * 32 + ((j < 8) ? (lane >> 4) * 8 + j : 16 + (lane >> 4) * 8 + (j - 8));
            Ub[kb][j] = (_Float16)u[K * kHD + N];
        }
    }

    // Load + project x rows into A-fragment layout, compute per-head norms.
#pragma unroll
    for (int rep = 0; rep < 4; ++rep) {
        const int item = rep * kP1Threads + tid;   // 1024 (row, head) items
        const int row  = item >> 6;
        const int head = item & 63;
        int kb, ln, j;
        frag_pos(row, head * 2, kb, ln, j);
        const float2 xv = *(const float2*)&x[(m0 + row) * kHD + head * 2];
        const float2 xp = project2(xv.x, xv.y);
        xfrag[kb][ln][j]     = (_Float16)xp.x;
        xfrag[kb][ln][j + 1] = (_Float16)xp.y;
        const float ex = xp.x + kEps, ey = xp.y + kEps;
        xnS[row][head] = fsqrt(ex * ex + ey * ey);
    }
    __syncthreads();

    // GEMM: 4 independent WMMAs per wave.
    const v16h A0 = *(const v16h*)(&xfrag[0][lane][0]);
    const v16h A1 = *(const v16h*)(&xfrag[1][lane][0]);
    const v16h A2 = *(const v16h*)(&xfrag[2][lane][0]);
    const v16h A3 = *(const v16h*)(&xfrag[3][lane][0]);
    const v8f vzero = {0.f, 0.f, 0.f, 0.f, 0.f, 0.f, 0.f, 0.f};
    v8f p0 = __builtin_amdgcn_wmma_f32_16x16x32_f16(false, A0, false, Ub[0],
                                                    (short)0, vzero, false, false);
    v8f p1 = __builtin_amdgcn_wmma_f32_16x16x32_f16(false, A1, false, Ub[1],
                                                    (short)0, vzero, false, false);
    v8f p2 = __builtin_amdgcn_wmma_f32_16x16x32_f16(false, A2, false, Ub[2],
                                                    (short)0, vzero, false, false);
    v8f p3 = __builtin_amdgcn_wmma_f32_16x16x32_f16(false, A3, false, Ub[3],
                                                    (short)0, vzero, false, false);
    {
        const int Mb = (lane >> 4) * 8;
#pragma unroll
        for (int r = 0; r < 8; ++r)
            UxS[Mb + r][N] = (p0[r] + p1[r]) + (p2[r] + p3[r]);
    }
    __syncthreads();

    // Per-head tail + writeout of Ux' into out[b,s,:].
#pragma unroll
    for (int rep = 0; rep < 4; ++rep) {
        const int item = rep * kP1Threads + tid;
        const int row  = item >> 6;
        const int head = item & 63;
        const float2 Uxp = mobmm_tail(UxS[row][head * 2], UxS[row][head * 2 + 1],
                                      xnS[row][head]);
        *(float2*)&out[(m0 + row) * kHD + head * 2] = Uxp;
    }
}

// ===========================================================================
// Phase 2: the serial scan. Only the W GEMM + one Mobius tail per step.
// ===========================================================================
__global__ __launch_bounds__(kP2Threads)
void hyper_rnn_scan_kernel(const float* __restrict__ w,
                           const float* __restrict__ b,
                           float* out) {           // aliased: Ux' in, h out
    __shared__ _Float16 hfrag[4][32][16];
    __shared__ float WhS0[kBT][kHD];   // partial sums, K-blocks 0-1
    __shared__ float WhS1[kBT][kHD];   // partial sums, K-blocks 2-3
    __shared__ float hnorm[kBT][kH];
    __shared__ float bsh[kHD];

    const int tid  = threadIdx.x;
    const int wave = tid >> 5;           // 0..15
    const int lane = tid & 31;
    const int b0   = blockIdx.x * kBT;

    // Wave -> (K-half, N-tile): waves 0-7 do K 0..63, waves 8-15 do K 64..127.
    const int  g   = wave >> 3;          // 0 or 1
    const int  nt  = wave & 7;
    const int  N   = nt * 16 + (lane & 15);

    // Register-resident W B-fragments for this wave's two K-blocks.
    v16h Wf[2];
#pragma unroll
    for (int kk = 0; kk < 2; ++kk) {
        const int kb = 2 * g + kk;
#pragma unroll
        for (int j = 0; j < 16; ++j) {
            const int K = kb * 32 + ((j < 8) ? (lane >> 4) * 8 + j : 16 + (lane >> 4) * 8 + (j - 8));
            Wf[kk][j] = (_Float16)w[K * kHD + N];
        }
    }
    if (tid < kHD) bsh[tid] = b[tid];

    float (*const accS)[kHD] = g ? WhS1 : WhS0;

    // Per-thread (row, head) item coordinates (2 items per thread).
    int rowI[2], headI[2], kbI[2], lnI[2], jI[2];
#pragma unroll
    for (int rep = 0; rep < 2; ++rep) {
        const int item = rep * kP2Threads + tid;   // 1024 (row, head) items
        rowI[rep]  = item >> 6;
        headI[rep] = item & 63;
        frag_pos(rowI[rep], headI[rep] * 2, kbI[rep], lnI[rep], jI[rep]);
    }

    // Init h0 = 0 fragments/norms; preload Ux'(t=0) into registers.
    float2 uxg[2];
#pragma unroll
    for (int rep = 0; rep < 2; ++rep) {
        hfrag[kbI[rep]][lnI[rep]][jI[rep]]     = (_Float16)0.0f;
        hfrag[kbI[rep]][lnI[rep]][jI[rep] + 1] = (_Float16)0.0f;
        hnorm[rowI[rep]][headI[rep]]           = 1.41421356e-15f;  // ||0+eps||
        uxg[rep] = *(const float2*)&out[((size_t)(b0 + rowI[rep]) * kS) * kHD +
                                        headI[rep] * 2];
    }
    __syncthreads();

    const v8f vzero = {0.f, 0.f, 0.f, 0.f, 0.f, 0.f, 0.f, 0.f};

    for (int t = 0; t < kS; ++t) {
        // ---- W GEMM: 2 independent WMMAs on this wave's K-half ----
        const v16h A0 = *(const v16h*)(&hfrag[2 * g + 0][lane][0]);
        const v16h A1 = *(const v16h*)(&hfrag[2 * g + 1][lane][0]);
        v8f p0 = __builtin_amdgcn_wmma_f32_16x16x32_f16(false, A0, false, Wf[0],
                                                        (short)0, vzero, false, false);
        v8f p1 = __builtin_amdgcn_wmma_f32_16x16x32_f16(false, A1, false, Wf[1],
                                                        (short)0, vzero, false, false);
        {
            const int Mb = (lane >> 4) * 8;
#pragma unroll
            for (int r = 0; r < 8; ++r) accS[Mb + r][N] = p0[r] + p1[r];
        }

        BAR_SIGNAL();
        BAR_WAIT();

        // ---- Elementwise: merge K-halves, one Mobius tail, two mob_adds ----
#pragma unroll
        for (int rep = 0; rep < 2; ++rep) {
            const int row = rowI[rep], head = headI[rep];
            const int kb = kbI[rep], ln = lnI[rep], j = jI[rep];

            const float whx = WhS0[row][head * 2]     + WhS1[row][head * 2];
            const float why = WhS0[row][head * 2 + 1] + WhS1[row][head * 2 + 1];
            const float2 Whp = mobmm_tail(whx, why, hnorm[row][head]);
            const float2 bb  = make_float2(bsh[head * 2], bsh[head * 2 + 1]);
            const float2 hn  = mob_add2(mob_add2(Whp, uxg[rep]), bb);

            // Overwrite Ux'(t) slot (already consumed into uxg last step).
            *(float2*)&out[((size_t)(b0 + row) * kS + t) * kHD + head * 2] = hn;

            // Stage h for step t+1.
            const float2 hp = project2(hn.x, hn.y);
            hfrag[kb][ln][j]     = (_Float16)hp.x;
            hfrag[kb][ln][j + 1] = (_Float16)hp.y;
            const float hx = hp.x + kEps, hy = hp.y + kEps;
            hnorm[row][head] = fsqrt(hx * hx + hy * hy);
        }

        BAR_SIGNAL();
        // --- barrier shadow: pipeline Ux'(t+1) global loads (no LDS traffic) ---
        if (t + 1 < kS) {
#pragma unroll
            for (int rep = 0; rep < 2; ++rep) {
                const float* uptr = &out[((size_t)(b0 + rowI[rep]) * kS + (t + 1)) * kHD +
                                         headI[rep] * 2];
                uxg[rep] = *(const float2*)uptr;
                if (t + 2 < kS) __builtin_prefetch(uptr + kHD, 0, 0);  // global_prefetch
            }
        }
        BAR_WAIT();
    }
}

extern "C" void kernel_launch(void* const* d_in, const int* in_sizes, int n_in,
                              void* d_out, int out_size, void* d_ws, size_t ws_size,
                              hipStream_t stream) {
    (void)in_sizes; (void)n_in; (void)d_ws; (void)ws_size; (void)out_size;
    const float* x = (const float*)d_in[0];
    const float* w = (const float*)d_in[1];
    const float* u = (const float*)d_in[2];
    const float* b = (const float*)d_in[3];
    float* out = (float*)d_out;

    // Phase 1: Ux' for all (b, s) — 16384 workgroups, chip-saturating WMMA.
    ux_precompute_kernel<<<(kB * kS) / kBT, kP1Threads, 0, stream>>>(x, u, out);
    // Phase 2: serial scan over S with only the W GEMM on the critical path.
    hyper_rnn_scan_kernel<<<kB / kBT, kP2Threads, 0, stream>>>(w, b, out);
}
